// SLSTM_27367531610424
// MI455X (gfx1250) — compile-verified, matrix operand order checked by reading, not run on hardware
//
#include <hip/hip_runtime.h>
#include <math.h>

// ---------------- problem constants ----------------
#define BATCH 32
#define SEQ   512
#define HDIM  256
#define EMBD  256
#define H3    768
#define H7    1792
#define H2    512
#define LBL   16
#define NROWS (BATCH * SEQ)   // 16384

typedef _Float16 half_t;
typedef _Float16 v16h __attribute__((ext_vector_type(16)));
typedef float    v8f  __attribute__((ext_vector_type(8)));

// ---------------- small helper kernels ----------------
__global__ void k_cvt_f16(const float* __restrict__ src, half_t* __restrict__ dst, int n) {
    int i = blockIdx.x * blockDim.x + threadIdx.x;
    if (i < n) dst[i] = (half_t)src[i];
}

__global__ void k_gather_emb(const int* __restrict__ tok, const float* __restrict__ E,
                             half_t* __restrict__ out) {
    int i = blockIdx.x * blockDim.x + threadIdx.x;
    if (i >= NROWS * EMBD) return;
    int n = i / EMBD, e = i - n * EMBD;
    out[i] = (half_t)E[(size_t)tok[n] * EMBD + e];
}

__global__ void k_init_hc(float* __restrict__ h, float* __restrict__ c,
                          const float* __restrict__ mask) {
    int i = blockIdx.x * blockDim.x + threadIdx.x;
    if (i >= NROWS * HDIM) return;
    float v = h[i] * mask[i / HDIM];
    h[i] = v; c[i] = v;
}

// mean over S for each (b,f); optionally duplicate into dst2
__global__ void k_colmean(const float* __restrict__ src, float* __restrict__ dst1,
                          float* __restrict__ dst2) {
    int i = blockIdx.x * blockDim.x + threadIdx.x;
    if (i >= BATCH * HDIM) return;
    int b = i / HDIM, f = i - b * HDIM;
    float s = 0.f;
    for (int t = 0; t < SEQ; ++t) s += src[(size_t)(b * SEQ + t) * HDIM + f];
    s *= (1.0f / SEQ);
    dst1[i] = s;
    if (dst2) dst2[i] = s;
}

// hcat[n] = [before(h), h, after(h)] in f16 (shift along seq within each batch)
__global__ void k_build_hcat(const float* __restrict__ h, half_t* __restrict__ hcat) {
    int i = blockIdx.x * blockDim.x + threadIdx.x;
    if (i >= NROWS * HDIM) return;
    int n = i / HDIM, f = i - n * HDIM;
    int s = n & (SEQ - 1);
    size_t base = (size_t)n * H3;
    hcat[base + HDIM + f]     = (half_t)h[i];
    hcat[base + f]            = (s > 0)       ? (half_t)h[i - HDIM] : (half_t)0.f;
    hcat[base + 2 * HDIM + f] = (s < SEQ - 1) ? (half_t)h[i + HDIM] : (half_t)0.f;
}

// tiny fp32 GEMM: C[M,N] = (acc?C:0) + A[M,K]@B[K,N] (+bias) (+sigmoid)
__global__ void k_smallmm(const float* __restrict__ A, const float* __restrict__ Bw,
                          const float* __restrict__ bias, float* __restrict__ C,
                          int M, int Nn, int K, int accumulate, int act) {
    int i = blockIdx.x * blockDim.x + threadIdx.x;
    if (i >= M * Nn) return;
    int m = i / Nn, n = i - m * Nn;
    float s = accumulate ? C[i] : 0.f;
    for (int k = 0; k < K; ++k) s += A[m * K + k] * Bw[k * Nn + n];
    if (bias) s += bias[n];
    if (act == 1) s = 1.f / (1.f + expf(-s));
    C[i] = s;
}

// SCell pointwise: gate nonlinearities + new_c/new_h (outputs pre-masked)
__global__ void k_scell(const float* __restrict__ G, const float* __restrict__ c,
                        const float* __restrict__ c_g, const float* __restrict__ mask,
                        float* __restrict__ hOut, float* __restrict__ cOut) {
    int i = blockIdx.x * blockDim.x + threadIdx.x;
    if (i >= NROWS * HDIM) return;
    int n = i / HDIM, f = i - n * HDIM;
    int b = n >> 9;              // S == 512
    int s = n & (SEQ - 1);
    size_t gb = (size_t)n * H7;
    float gi = G[gb + 0 * HDIM + f];
    float gl = G[gb + 1 * HDIM + f];
    float gr = G[gb + 2 * HDIM + f];
    float gf = G[gb + 3 * HDIM + f];
    float gs = G[gb + 4 * HDIM + f];
    float go = G[gb + 5 * HDIM + f];          // stays raw (as in source)
    float gu = tanhf(G[gb + 6 * HDIM + f]);
    float si = 1.f / (1.f + expf(-gi));
    float sl = 1.f / (1.f + expf(-gl));
    float sr = 1.f / (1.f + expf(-gr));
    float sf = 1.f / (1.f + expf(-gf));
    float ss = 1.f / (1.f + expf(-gs));
    float mx = fmaxf(fmaxf(fmaxf(si, sl), fmaxf(sr, sf)), ss);
    float ei = expf(si - mx), el = expf(sl - mx), er = expf(sr - mx);
    float ef = expf(sf - mx), es = expf(ss - mx);
    float inv = 1.f / (ei + el + er + ef + es);
    float cb = (s > 0)       ? c[i - HDIM] : 0.f;
    float ca = (s < SEQ - 1) ? c[i + HDIM] : 0.f;
    float cc = c[i];
    float nc = cb * (el * inv) + cc * (ef * inv) + (er * inv) * ca
             + (es * inv) * c_g[b * HDIM + f] + (ei * inv) * gu;
    float nh = go * tanhf(nc);
    float mk = mask[n];
    cOut[i] = nc * mk;
    hOut[i] = nh * mk;
}

// GCell: f_w = softmax_S(sigmoid(fwlog) - (1-mask)*1e16); update g, c_g in place
__global__ void k_gcell(const float* __restrict__ fwlog, const float* __restrict__ mask,
                        const float* __restrict__ c, const float* __restrict__ fo,
                        float* __restrict__ g, float* __restrict__ c_g) {
    int i = blockIdx.x * blockDim.x + threadIdx.x;
    if (i >= BATCH * HDIM) return;
    int b = i / HDIM, f = i - b * HDIM;
    float mx = -3.0e38f;
    for (int s = 0; s < SEQ; ++s) {
        float lv = fwlog[(size_t)(b * SEQ + s) * HDIM + f];
        float x = 1.f / (1.f + expf(-lv)) - (1.f - mask[b * SEQ + s]) * 1e16f;
        mx = fmaxf(mx, x);
    }
    float sum = 0.f;
    for (int s = 0; s < SEQ; ++s) {
        float lv = fwlog[(size_t)(b * SEQ + s) * HDIM + f];
        float x = 1.f / (1.f + expf(-lv)) - (1.f - mask[b * SEQ + s]) * 1e16f;
        sum += expf(x - mx);
    }
    float inv = 1.f / sum, acc = 0.f;
    for (int s = 0; s < SEQ; ++s) {
        float lv = fwlog[(size_t)(b * SEQ + s) * HDIM + f];
        float x = 1.f / (1.f + expf(-lv)) - (1.f - mask[b * SEQ + s]) * 1e16f;
        acc += c[(size_t)(b * SEQ + s) * HDIM + f] * (expf(x - mx) * inv);
    }
    float fg = fo[b * H2 + f];
    float og = fo[b * H2 + HDIM + f];
    float ncg = fg * c_g[i] + acc;
    c_g[i] = ncg;
    g[i] = og * tanhf(ncg);
}

__global__ void k_gmean(const float* __restrict__ g, float* __restrict__ out) {
    int f = blockIdx.x * blockDim.x + threadIdx.x;
    if (f >= HDIM) return;
    float s = 0.f;
    for (int b = 0; b < BATCH; ++b) s += g[b * HDIM + f];
    out[f] = s * (1.f / BATCH);
}

// ---------------- WMMA GEMM ----------------
// C[M,Nn] f32 = A[M,K] f16 (row-major, lda) @ Bw[K,Nn] f16 (row-major)
//   + optional addFull[M,Nn] + optional addBatch[row/S, Nn] + optional bias[Nn]
// Block tile 128x64, 256 threads (8 waves); each wave owns a 32x32 sub-tile:
// 2 A-fragments x 2 B-fragments -> 4 v_wmma_f32_16x16x32_f16 per K-step.
__global__ __launch_bounds__(256)
void k_wmma_gemm(const half_t* __restrict__ A, int lda,
                 const half_t* __restrict__ Bw,
                 float* __restrict__ C,
                 const float* __restrict__ addFull,
                 const float* __restrict__ addBatch,
                 const float* __restrict__ bias,
                 int Nn, int K) {
    __shared__ __align__(16) half_t As [128][40];  // 32 K-halves + 8 pad (16B-aligned rows)
    __shared__ __align__(16) half_t Bts[64][40];   // B staged transposed: [n][k]

    const int nBlocksN = Nn / 64;
    const int bm = (blockIdx.x / nBlocksN) * 128;
    const int bn = (blockIdx.x % nBlocksN) * 64;

    const int tid  = threadIdx.x;
    const int wid  = tid >> 5;
    const int lane = tid & 31;
    const int wm   = wid & 3;           // 4 wave-rows of 32
    const int wn   = wid >> 2;          // 2 wave-cols of 32

    const int mfrag = lane & 15;
    const int kbase = (lane >> 4) << 3; // 0 or 8 (ISA 16-bit matrix lane layout)
    const int kq    = kbase >> 3;       // uint4 index within LDS row

    v8f acc00 = {}, acc01 = {}, acc10 = {}, acc11 = {};

    // staging thread mapping (A): each thread loads 2 consecutive uint4 of one row
    const int ar = tid >> 1;            // 0..127
    const int ac = (tid & 1) * 2;       // uint4 chunk 0 or 2
    // staging thread mapping (B): 2(K) x 4(N) micro-tile per thread
    const int bkp = tid >> 4;           // 0..15 -> K rows 2*bkp, 2*bkp+1
    const int bcg = tid & 15;           // 0..15 -> cols bcg*4 .. +3

    for (int k0 = 0; k0 < K; k0 += 32) {
        // ---- stage A tile 128x32 ----
        {
            const uint4* gp = reinterpret_cast<const uint4*>(
                A + (size_t)(bm + ar) * lda + k0) + ac;
            uint4* sp = reinterpret_cast<uint4*>(&As[ar][0]) + ac;
            sp[0] = gp[0];
            sp[1] = gp[1];
            if (k0 + 32 < K)
                __builtin_prefetch(A + (size_t)(bm + ar) * lda + k0 + 32 + ac * 8, 0, 1);
        }
        // ---- stage B tile 32x64, transposed into LDS with packed b32 stores ----
        {
            union { uint2 q; half_t h[4]; } r0, r1;
            const half_t* bp = Bw + (size_t)(k0 + 2 * bkp) * Nn + bn + bcg * 4;
            r0.q = *reinterpret_cast<const uint2*>(bp);
            r1.q = *reinterpret_cast<const uint2*>(bp + Nn);
#pragma unroll
            for (int t = 0; t < 4; ++t) {
                union { unsigned int u; half_t h[2]; } pk;
                pk.h[0] = r0.h[t];
                pk.h[1] = r1.h[t];
                *reinterpret_cast<unsigned int*>(&Bts[bcg * 4 + t][2 * bkp]) = pk.u;
            }
            if (k0 + 32 < K)
                __builtin_prefetch(Bw + (size_t)(k0 + 32 + 2 * bkp) * Nn + bn + bcg * 4, 0, 1);
        }
        __syncthreads();

        union Frag { uint4 q[2]; v16h v; };
        Frag a0, a1, b0, b1;
        {
            const uint4* rp = reinterpret_cast<const uint4*>(&As[wm * 32 + mfrag][0]);
            a0.q[0] = rp[kq];
            a0.q[1] = rp[kq + 2];
            const uint4* rq = reinterpret_cast<const uint4*>(&As[wm * 32 + 16 + mfrag][0]);
            a1.q[0] = rq[kq];
            a1.q[1] = rq[kq + 2];
        }
        {
            const uint4* rp = reinterpret_cast<const uint4*>(&Bts[wn * 32 + mfrag][0]);
            b0.q[0] = rp[kq];
            b0.q[1] = rp[kq + 2];
            const uint4* rq = reinterpret_cast<const uint4*>(&Bts[wn * 32 + 16 + mfrag][0]);
            b1.q[0] = rq[kq];
            b1.q[1] = rq[kq + 2];
        }
        acc00 = __builtin_amdgcn_wmma_f32_16x16x32_f16(false, a0.v, false, b0.v,
                                                       (short)0, acc00, false, false);
        acc01 = __builtin_amdgcn_wmma_f32_16x16x32_f16(false, a0.v, false, b1.v,
                                                       (short)0, acc01, false, false);
        acc10 = __builtin_amdgcn_wmma_f32_16x16x32_f16(false, a1.v, false, b0.v,
                                                       (short)0, acc10, false, false);
        acc11 = __builtin_amdgcn_wmma_f32_16x16x32_f16(false, a1.v, false, b1.v,
                                                       (short)0, acc11, false, false);
        __syncthreads();
    }

    // epilogue: D layout = vgpr r -> M=r(+8 for lanes 16..31), N = lane&15
    const int mofs = (lane >> 4) * 8;
    const int ncol = lane & 15;
#pragma unroll
    for (int mt = 0; mt < 2; ++mt) {
        v8f* accA = mt ? &acc10 : &acc00;
        v8f* accB = mt ? &acc11 : &acc01;
#pragma unroll
        for (int r = 0; r < 8; ++r) {
            int grow  = bm + wm * 32 + mt * 16 + mofs + r;
            int gcol  = bn + wn * 32 + ncol;
            size_t i0 = (size_t)grow * Nn + gcol;
            size_t i1 = i0 + 16;
            float e0 = (*accA)[r], e1 = (*accB)[r];
            if (bias)    { e0 += bias[gcol];          e1 += bias[gcol + 16]; }
            if (addFull) { e0 += addFull[i0];         e1 += addFull[i1]; }
            if (addBatch){ int bb = grow >> 9;        // row / S
                           e0 += addBatch[(size_t)bb * Nn + gcol];
                           e1 += addBatch[(size_t)bb * Nn + gcol + 16]; }
            C[i0] = e0;
            C[i1] = e1;
        }
    }
}

// ---------------- launcher ----------------
extern "C" void kernel_launch(void* const* d_in, const int* in_sizes, int n_in,
                              void* d_out, int out_size, void* d_ws, size_t ws_size,
                              hipStream_t stream) {
    (void)in_sizes; (void)n_in; (void)out_size; (void)ws_size;

    const int*   node_text = (const int*)  d_in[2];
    const float* word_mask = (const float*)d_in[3];
    const float* E   = (const float*)d_in[7];
    const float* Wp  = (const float*)d_in[8];
    const float* bp  = (const float*)d_in[9];
    const float* Ws  = (const float*)d_in[10];
    const float* Us  = (const float*)d_in[11];
    const float* Vs  = (const float*)d_in[12];
    const float* bVs = (const float*)d_in[13];
    const float* WgW = (const float*)d_in[14];
    const float* Wgw = (const float*)d_in[15];
    const float* UgU = (const float*)d_in[16];
    const float* bgU = (const float*)d_in[17];
    const float* ugu = (const float*)d_in[18];
    const float* bgu = (const float*)d_in[19];
    const float* Wo  = (const float*)d_in[20];
    const float* bo  = (const float*)d_in[21];
    float* out = (float*)d_out;

    char* w = (char*)d_ws;
    size_t off = 0;
    auto take = [&](size_t bytes) -> void* {
        void* p = w + off;
        off = (off + bytes + 255) & ~(size_t)255;
        return p;
    };

    half_t* embf16 = (half_t*)take((size_t)NROWS * EMBD * 2);
    half_t* Wp16   = (half_t*)take((size_t)EMBD * HDIM * 2);
    half_t* Ws16   = (half_t*)take((size_t)H3 * H7 * 2);
    half_t* Us16   = (half_t*)take((size_t)EMBD * H7 * 2);
    half_t* ugu16  = (half_t*)take((size_t)HDIM * HDIM * 2);
    float*  xU     = (float*)take((size_t)NROWS * H7 * 4);
    float*  gates  = (float*)take((size_t)NROWS * H7 * 4);
    float*  hA     = (float*)take((size_t)NROWS * HDIM * 4);
    float*  hB     = (float*)take((size_t)NROWS * HDIM * 4);
    float*  cA     = (float*)take((size_t)NROWS * HDIM * 4);
    float*  cB     = (float*)take((size_t)NROWS * HDIM * 4);
    half_t* hcat   = (half_t*)take((size_t)NROWS * H3 * 2);
    float*  fwlog  = (float*)take((size_t)NROWS * HDIM * 4);
    float*  gbuf   = (float*)take((size_t)BATCH * HDIM * 4);
    float*  cgbuf  = (float*)take((size_t)BATCH * HDIM * 4);
    float*  havg   = (float*)take((size_t)BATCH * HDIM * 4);
    float*  gwb    = (float*)take((size_t)BATCH * HDIM * 4);
    float*  gvec   = (float*)take((size_t)BATCH * H7 * 4);
    float*  fob    = (float*)take((size_t)BATCH * H2 * 4);
    float*  gmean  = (float*)take((size_t)HDIM * 4);

    const int TPB = 256;
    auto blocks = [](long long n) { return (int)((n + 255) / 256); };

    // weight + embedding prep
    k_cvt_f16<<<blocks(EMBD * HDIM), TPB, 0, stream>>>(Wp, Wp16, EMBD * HDIM);
    k_cvt_f16<<<blocks((long long)H3 * H7), TPB, 0, stream>>>(Ws, Ws16, H3 * H7);
    k_cvt_f16<<<blocks(EMBD * H7), TPB, 0, stream>>>(Us, Us16, EMBD * H7);
    k_cvt_f16<<<blocks(HDIM * HDIM), TPB, 0, stream>>>(ugu, ugu16, HDIM * HDIM);
    k_gather_emb<<<blocks((long long)NROWS * EMBD), TPB, 0, stream>>>(node_text, E, embf16);

    // proj = emb @ Wp + bp  -> hA
    k_wmma_gemm<<<(NROWS / 128) * (HDIM / 64), TPB, 0, stream>>>(
        embf16, EMBD, Wp16, hA, nullptr, nullptr, bp, HDIM, EMBD);
    k_init_hc<<<blocks((long long)NROWS * HDIM), TPB, 0, stream>>>(hA, cA, word_mask);
    k_colmean<<<blocks(BATCH * HDIM), TPB, 0, stream>>>(hA, gbuf, cgbuf);
    k_build_hcat<<<blocks((long long)NROWS * HDIM), TPB, 0, stream>>>(hA, hcat);

    // xU = emb @ Us (constant across layers)
    k_wmma_gemm<<<(NROWS / 128) * (H7 / 64), TPB, 0, stream>>>(
        embf16, EMBD, Us16, xU, nullptr, nullptr, nullptr, H7, EMBD);

    float* hCur = hA; float* cCur = cA; float* hNxt = hB; float* cNxt = cB;
    for (int layer = 0; layer < 3; ++layer) {
        // gvec = g @ Vs + bVs   [32,1792]
        k_smallmm<<<blocks(BATCH * H7), TPB, 0, stream>>>(
            gbuf, Vs, bVs, gvec, BATCH, H7, HDIM, 0, 0);
        // gates = hcat @ Ws + xU + gvec(broadcast per batch)
        k_wmma_gemm<<<(NROWS / 128) * (H7 / 64), TPB, 0, stream>>>(
            hcat, H3, Ws16, gates, xU, gvec, nullptr, H7, H3);
        // h_avg
        k_colmean<<<blocks(BATCH * HDIM), TPB, 0, stream>>>(hCur, havg, nullptr);
        // fo = sigmoid(g@WgW + h_avg@UgU + bgU)
        k_smallmm<<<blocks(BATCH * H2), TPB, 0, stream>>>(
            gbuf, WgW, nullptr, fob, BATCH, H2, HDIM, 0, 0);
        k_smallmm<<<blocks(BATCH * H2), TPB, 0, stream>>>(
            havg, UgU, bgU, fob, BATCH, H2, HDIM, 1, 1);
        // gw = g @ Wgw
        k_smallmm<<<blocks(BATCH * HDIM), TPB, 0, stream>>>(
            gbuf, Wgw, nullptr, gwb, BATCH, HDIM, HDIM, 0, 0);
        // fwlog = h @ ugu + bgu + gw(broadcast)   (A = center block of hcat)
        k_wmma_gemm<<<(NROWS / 128) * (HDIM / 64), TPB, 0, stream>>>(
            hcat + HDIM, H3, ugu16, fwlog, nullptr, gwb, bgu, HDIM, HDIM);
        // SCell pointwise -> hNxt, cNxt (masked)
        k_scell<<<blocks((long long)NROWS * HDIM), TPB, 0, stream>>>(
            gates, cCur, cgbuf, word_mask, hNxt, cNxt);
        // GCell: softmax over S + g / c_g update (uses pre-update h,c)
        k_gcell<<<blocks(BATCH * HDIM), TPB, 0, stream>>>(
            fwlog, word_mask, cCur, fob, gbuf, cgbuf);
        // swap, rebuild hcat from new h
        float* t;
        t = hCur; hCur = hNxt; hNxt = t;
        t = cCur; cCur = cNxt; cNxt = t;
        k_build_hcat<<<blocks((long long)NROWS * HDIM), TPB, 0, stream>>>(hCur, hcat);
    }

    // heads: (g @ Wo + bo) [32,16]  and  (mean_B(g) @ Wo + bo) [16]
    k_smallmm<<<blocks(BATCH * LBL), TPB, 0, stream>>>(
        gbuf, Wo, bo, out, BATCH, LBL, HDIM, 0, 0);
    k_gmean<<<1, TPB, 0, stream>>>(gbuf, gmean);
    k_smallmm<<<1, 64, 0, stream>>>(gmean, Wo, bo, out + BATCH * LBL, 1, LBL, HDIM, 0, 0);
}
